// BiLSTMCRFTagger_62612033241797
// MI455X (gfx1250) — matrix operand order repaired and checked by reading, not compiled, not run on hardware
//
#include <hip/hip_runtime.h>
#include <hip/hip_bf16.h>

typedef __attribute__((ext_vector_type(16))) __bf16 v16bf;
typedef __attribute__((ext_vector_type(8)))  __bf16 v8bf;
typedef __attribute__((ext_vector_type(8)))  float  v8f;

union ABu { v16bf v; v8bf h[2]; };

#define WMMA_BF16(a,b,c) \
  __builtin_amdgcn_wmma_f32_16x16x32_bf16(false,(a),false,(b),(short)0,(c),false,false)

#define Bn   32
#define Tn   256
#define En   300
#define Ep   320
#define Hn   512
#define G4H  2048
#define Ln   17
#define BT   (Bn*Tn)

// ---------------- prep kernels ----------------

// convert f32 [rows x ksrc] -> bf16 [rows x kdst], zero-padding k >= ksrc
__global__ void k_conv_pad(const float* __restrict__ src, __bf16* __restrict__ dst,
                           int rows, int ksrc, int kdst) {
  int idx = blockIdx.x * blockDim.x + threadIdx.x;
  if (idx >= rows * kdst) return;
  int r = idx / kdst, k = idx - r * kdst;
  float v = (k < ksrc) ? src[(size_t)r * ksrc + k] : 0.f;
  dst[idx] = (__bf16)v;
}

__global__ void k_bias(const float* __restrict__ bihf, const float* __restrict__ bhhf,
                       const float* __restrict__ bihb, const float* __restrict__ bhhb,
                       float* __restrict__ dst) {
  int i = blockIdx.x * blockDim.x + threadIdx.x;
  if (i < G4H)          dst[i] = bihf[i] + bhhf[i];
  else if (i < 2 * G4H) dst[i] = bihb[i - G4H] + bhhb[i - G4H];
}

// embedding gather -> bf16 [BT x Ep]
__global__ void k_gather(const int* __restrict__ ids, const float* __restrict__ emb,
                         __bf16* __restrict__ xb) {
  int idx = blockIdx.x * blockDim.x + threadIdx.x;
  if (idx >= BT * Ep) return;
  int r = idx / Ep, k = idx - r * Ep;
  int id = ids[r];
  float v = (k < En) ? emb[(size_t)id * En + k] : 0.f;
  xb[idx] = (__bf16)v;
}

// ---------------- input projection GEMM: xg[dir] = xb @ wih[dir]^T + bias ----------------
// grid (N/64, M/128, 2), block 256 (8 waves); each wave: 16(M) x 64(N) tile, K=320.
__global__ __launch_bounds__(256)
void k_gemm_xg(const __bf16* __restrict__ xb, const __bf16* __restrict__ wih,
               const float* __restrict__ bias, float* __restrict__ xg) {
  int lane = threadIdx.x & 31, wv = threadIdx.x >> 5;
  int laneLo = lane & 15, laneHi = lane >> 4;
  int dir = blockIdx.z;
  const __bf16* W  = wih  + (size_t)dir * G4H * Ep;
  const float*  Bv = bias + (size_t)dir * G4H;
  float*        XG = xg   + (size_t)dir * BT * G4H;
  int m0 = blockIdx.y * 128 + wv * 16;
  int n0 = blockIdx.x * 64;

  v8f acc[4];
#pragma unroll
  for (int ni = 0; ni < 4; ++ni)
#pragma unroll
    for (int v = 0; v < 8; ++v) acc[ni][v] = 0.f;

  for (int kb = 0; kb < Ep; kb += 32) {
    ABu a;
    const __bf16* ap = xb + (size_t)(m0 + laneLo) * Ep + kb + 8 * laneHi;
    a.h[0] = *(const v8bf*)ap;
    a.h[1] = *(const v8bf*)(ap + 16);
#pragma unroll
    for (int ni = 0; ni < 4; ++ni) {
      ABu b;
      const __bf16* bp = W + (size_t)(n0 + 16 * ni + laneLo) * Ep + kb + 8 * laneHi;
      b.h[0] = *(const v8bf*)bp;
      b.h[1] = *(const v8bf*)(bp + 16);
      acc[ni] = WMMA_BF16(a.v, b.v, acc[ni]);
    }
  }
#pragma unroll
  for (int ni = 0; ni < 4; ++ni) {
    int n = n0 + 16 * ni + laneLo;
    float bv = Bv[n];
#pragma unroll
    for (int v = 0; v < 8; ++v) {
      int m = m0 + 8 * laneHi + v;
      XG[(size_t)m * G4H + n] = acc[ni][v] + bv;
    }
  }
}

// ---------------- persistent LSTM scan: one workgroup per direction ----------------
// 1024 threads = 32 waves; wave w owns hidden units [16w,16w+16); its 4 N-tiles are
// the 4 gates (n = g*512 + u), so i/f/g/o for (batch,unit) live in the same lane.
__global__ __launch_bounds__(1024)
void k_lstm(const float* __restrict__ xg_all, const __bf16* __restrict__ whh_all,
            float* __restrict__ h_all) {
  int dir = blockIdx.x;
  const float*  XG = xg_all  + (size_t)dir * BT * G4H;
  const __bf16* W  = whh_all + (size_t)dir * G4H * Hn;
  float*        HO = h_all   + (size_t)dir * BT * Hn;

  __shared__ __align__(16) __bf16 hbuf[2][Bn][Hn];   // 64 KB double buffer

  int lane = threadIdx.x & 31, wv = threadIdx.x >> 5;
  int laneLo = lane & 15, laneHi = lane >> 4;
  int u = wv * 16 + laneLo;                           // hidden unit owned by this lane

  for (int i = threadIdx.x; i < Bn * Hn; i += 1024)
    ((__bf16*)hbuf[0])[i] = (__bf16)0.f;
  __syncthreads();

  float cst[2][8];
#pragma unroll
  for (int mi = 0; mi < 2; ++mi)
#pragma unroll
    for (int v = 0; v < 8; ++v) cst[mi][v] = 0.f;

  for (int t = 0; t < Tn; ++t) {
    int p = dir ? (Tn - 1 - t) : t;
    int cur = t & 1, nxt = cur ^ 1;

    v8f acc[2][4];
#pragma unroll
    for (int mi = 0; mi < 2; ++mi)
#pragma unroll
      for (int g = 0; g < 4; ++g) {
        int n = g * Hn + u;
#pragma unroll
        for (int v = 0; v < 8; ++v) {
          int m = 16 * mi + 8 * laneHi + v;
          acc[mi][g][v] = XG[((size_t)m * Tn + p) * G4H + n];
        }
      }

    for (int kb = 0; kb < Hn; kb += 32) {
      ABu a0, a1;
      const __bf16* ap0 = &hbuf[cur][laneLo][kb + 8 * laneHi];
      a0.h[0] = *(const v8bf*)ap0;
      a0.h[1] = *(const v8bf*)(ap0 + 16);
      const __bf16* ap1 = &hbuf[cur][16 + laneLo][kb + 8 * laneHi];
      a1.h[0] = *(const v8bf*)ap1;
      a1.h[1] = *(const v8bf*)(ap1 + 16);
#pragma unroll
      for (int g = 0; g < 4; ++g) {
        ABu b;
        const __bf16* bp = W + (size_t)(g * Hn + u) * Hn + kb + 8 * laneHi;
        b.h[0] = *(const v8bf*)bp;
        b.h[1] = *(const v8bf*)(bp + 16);
        acc[0][g] = WMMA_BF16(a0.v, b.v, acc[0][g]);
        acc[1][g] = WMMA_BF16(a1.v, b.v, acc[1][g]);
      }
    }

#pragma unroll
    for (int mi = 0; mi < 2; ++mi)
#pragma unroll
      for (int v = 0; v < 8; ++v) {
        float iv = 1.f / (1.f + __expf(-acc[mi][0][v]));
        float fv = 1.f / (1.f + __expf(-acc[mi][1][v]));
        float gv = tanhf(acc[mi][2][v]);
        float ov = 1.f / (1.f + __expf(-acc[mi][3][v]));
        float c  = fv * cst[mi][v] + iv * gv;
        cst[mi][v] = c;
        float h  = ov * tanhf(c);
        int m = 16 * mi + 8 * laneHi + v;
        hbuf[nxt][m][u] = (__bf16)h;
        HO[((size_t)m * Tn + p) * Hn + u] = h;
      }
    __syncthreads();
  }
}

// ---------------- classifier: em[b,t,l] = [hf|hb] . w_cls[l] + b_cls[l] ----------------
__global__ __launch_bounds__(256)
void k_cls(const float* __restrict__ h_all, const float* __restrict__ wcls,
           const float* __restrict__ bcls, float* __restrict__ em) {
  int lane = threadIdx.x & 31, wv = threadIdx.x >> 5;
  int gid = blockIdx.x * 8 + wv;                       // b*Tn + t
  const float* hf = h_all + (size_t)gid * Hn;
  const float* hb = h_all + (size_t)BT * Hn + (size_t)gid * Hn;
  if (lane >= Ln) return;
  float s = bcls[lane];
  const float* wr = wcls + (size_t)lane * (2 * Hn);
  for (int k = 0; k < Hn; ++k) s += hf[k] * wr[k];
  for (int k = 0; k < Hn; ++k) s += hb[k] * wr[Hn + k];
  em[(size_t)gid * Ln + lane] = s;
}

// ---------------- CRF loss: one wave per batch row, lane j holds alpha_j ----------------
__global__ __launch_bounds__(1024)
void k_crf(const float* __restrict__ em, const int* __restrict__ labels,
           const unsigned char* __restrict__ mask, const float* __restrict__ start_t,
           const float* __restrict__ end_t, const float* __restrict__ trans,
           float* __restrict__ out) {
  __shared__ float sT[Ln * Ln], sS[Ln], sE[Ln], sL[Bn];
  int tid = threadIdx.x;
  if (tid < Ln * Ln) sT[tid] = trans[tid];
  if (tid < Ln) { sS[tid] = start_t[tid]; sE[tid] = end_t[tid]; }
  __syncthreads();

  int lane = tid & 31, b = tid >> 5;
  int j = (lane < Ln) ? lane : 0;
  const float* EM = em + (size_t)b * Tn * Ln;
  const int* LB = labels + b * Tn;
  const unsigned char* MK = mask + b * Tn;

  float alpha = (lane < Ln) ? (sS[j] + EM[j]) : -3.0e38f;
  int tp = LB[0];
  float score = sS[tp] + EM[tp];                       // redundantly on all lanes

  for (int t = 1; t < Tn; ++t) {
    float mt = MK[t] ? 1.f : 0.f;
    int tt = LB[t];
    score += mt * (sT[tp * Ln + tt] + EM[(size_t)t * Ln + tt]);
    if (MK[t]) tp = tt;

    float mx = -3.0e38f;
    for (int i = 0; i < Ln; ++i) {
      float ai = __shfl(alpha, i, 32);
      mx = fmaxf(mx, ai + sT[i * Ln + j]);
    }
    float sm = 0.f;
    for (int i = 0; i < Ln; ++i) {
      float ai = __shfl(alpha, i, 32);
      sm += __expf(ai + sT[i * Ln + j] - mx);
    }
    float e = (lane < Ln) ? EM[(size_t)t * Ln + j] : 0.f;
    float nx = mx + __logf(sm) + e;
    if (mt > 0.5f && lane < Ln) alpha = nx;
  }
  score += sE[tp];

  float mx = -3.0e38f;
  for (int i = 0; i < Ln; ++i) { float v = __shfl(alpha, i, 32) + sE[i]; mx = fmaxf(mx, v); }
  float sm = 0.f;
  for (int i = 0; i < Ln; ++i) { float v = __shfl(alpha, i, 32) + sE[i]; sm += __expf(v - mx); }
  float logZ = mx + __logf(sm);

  if (lane == 0) sL[b] = score - logZ;
  __syncthreads();
  if (tid == 0) {
    float s = 0.f;
    for (int bb = 0; bb < Bn; ++bb) s += sL[bb];
    out[0] = -s / (float)Bn;
  }
}

// ---------------- host launcher ----------------
extern "C" void kernel_launch(void* const* d_in, const int* in_sizes, int n_in,
                              void* d_out, int out_size, void* d_ws, size_t ws_size,
                              hipStream_t stream) {
  (void)in_sizes; (void)n_in; (void)out_size; (void)ws_size;
  const int*   ids    = (const int*)  d_in[0];
  const int*   labels = (const int*)  d_in[1];
  const unsigned char* mask = (const unsigned char*)d_in[2];
  const float* emb  = (const float*)d_in[3];
  const float* wihf = (const float*)d_in[4];
  const float* whhf = (const float*)d_in[5];
  const float* bihf = (const float*)d_in[6];
  const float* bhhf = (const float*)d_in[7];
  const float* wihb = (const float*)d_in[8];
  const float* whhb = (const float*)d_in[9];
  const float* bihb = (const float*)d_in[10];
  const float* bhhb = (const float*)d_in[11];
  const float* wcls = (const float*)d_in[12];
  const float* bcls = (const float*)d_in[13];
  const float* st   = (const float*)d_in[14];
  const float* et   = (const float*)d_in[15];
  const float* tr   = (const float*)d_in[16];

  char* ws = (char*)d_ws;
  __bf16* xb   = (__bf16*)(ws + 0);           //  8192*320 bf16      = 5,242,880 B
  __bf16* wih  = (__bf16*)(ws + 5242880);     //  2*2048*320 bf16    = 2,621,440 B
  __bf16* whh  = (__bf16*)(ws + 7864320);     //  2*2048*512 bf16    = 4,194,304 B
  float*  bias = (float*) (ws + 12058624);    //  2*2048 f32         = 16,384 B
  float*  xg   = (float*) (ws + 12075008);    //  2*8192*2048 f32    = 134,217,728 B
  float*  hbuf = (float*) (ws + 146292736);   //  2*8192*512 f32     = 33,554,432 B
  float*  em   = (float*) (ws + 179847168);   //  8192*17 f32        = 557,056 B

  k_conv_pad<<<(2048 * 320 + 255) / 256, 256, 0, stream>>>(wihf, wih, 2048, 300, 320);
  k_conv_pad<<<(2048 * 320 + 255) / 256, 256, 0, stream>>>(wihb, wih + (size_t)2048 * 320, 2048, 300, 320);
  k_conv_pad<<<(2048 * 512 + 255) / 256, 256, 0, stream>>>(whhf, whh, 2048, 512, 512);
  k_conv_pad<<<(2048 * 512 + 255) / 256, 256, 0, stream>>>(whhb, whh + (size_t)2048 * 512, 2048, 512, 512);
  k_bias<<<(2 * 2048 + 255) / 256, 256, 0, stream>>>(bihf, bhhf, bihb, bhhb, bias);
  k_gather<<<(BT * Ep + 255) / 256, 256, 0, stream>>>(ids, emb, xb);

  dim3 gg(32, 64, 2);                                  // N/64, M/128, dir
  k_gemm_xg<<<gg, 256, 0, stream>>>(xb, wih, bias, xg);

  k_lstm<<<2, 1024, 0, stream>>>(xg, whh, hbuf);       // 1 persistent WG per direction

  k_cls<<<1024, 256, 0, stream>>>(hbuf, wcls, bcls, em);

  k_crf<<<1, 1024, 0, stream>>>(em, labels, mask, st, et, tr, (float*)d_out);
}
